// MHDPA_55327768708764
// MI455X (gfx1250) — matrix operand order repaired and checked by reading, not compiled, not run on hardware
//
#include <hip/hip_runtime.h>
#include <hip/hip_fp16.h>

// ---------------------------------------------------------------------------
// MHDPA with softmax over the QUERY axis (per key column).
//   q,k,v = LN(x^T W)          [B,F,I] f16 (v stored transposed [B,I,F])
//   pass A: m[b,k]=max_q s, Z[b,k]=sum_q exp(s-m)  via WMMA QK^T sweep
//   pass B: out = (exp(S-m)/Z) V  via WMMA, S^T tiles feed P@V directly
// B=16 D=37 F=4096 I=64.  QK^T and PV use v_wmma_f32_16x16x32_f16; projections are scalar FMA.
// Workspace use: qh(8MB) kh(8MB) vt(8MB) m(256KB) z(256KB) ~= 24.5 MB.
// ---------------------------------------------------------------------------

typedef __attribute__((ext_vector_type(16))) _Float16 v16h;
typedef __attribute__((ext_vector_type(8)))  _Float16 v8h;
typedef __attribute__((ext_vector_type(8)))  float    v8f;

#define B_ 16
#define D_ 37
#define F_ 4096
#define I_ 64

static __device__ __forceinline__ v8f wmma16(v16h a, v16h b, v8f c) {
  // v_wmma_f32_16x16x32_f16: D = A(16x32 f16) x B(32x16 f16) + C(16x16 f32)
  return __builtin_amdgcn_wmma_f32_16x16x32_f16(false, a, false, b, (short)0, c,
                                                false, false);
}

// A-operand (16x32 f16, M=row): lanes 0-15 hold K=0..7 (v0-3) and K=16..23
// (v4-7); lanes 16-31 hold K=8..15 and K=24..31. p points at the row.
static __device__ __forceinline__ v16h load_A16(const _Float16* p, int lane) {
  const int off = (lane & 16) ? 8 : 0;
  v8h x0 = *(const v8h*)(p + off);
  v8h x1 = *(const v8h*)(p + 16 + off);
  return __builtin_shufflevector(x0, x1, 0, 1, 2, 3, 4, 5, 6, 7,
                                 8, 9, 10, 11, 12, 13, 14, 15);
}

// B-operand (32x16 f16, N=col): lanes 0-15 hold K=0..7 and K=16..23,
// lanes 16-31 K=8..15 and K=24..31.
// p points at the column's contiguous K-major data.
static __device__ __forceinline__ v16h load_B16(const _Float16* p, int lane) {
  const int off = (lane & 16) ? 8 : 0;
  v8h x0 = *(const v8h*)(p + off);
  v8h x1 = *(const v8h*)(p + 16 + off);
  return __builtin_shufflevector(x0, x1, 0, 1, 2, 3, 4, 5, 6, 7,
                                 8, 9, 10, 11, 12, 13, 14, 15);
}

// ---------------------------------------------------------------------------
// Kernel 1: projections + parameter-free LayerNorm, emit f16 q/k and v^T.
// One thread per (b, f) row; W matrices cached in LDS (3 x 37 x 64 f32).
// ---------------------------------------------------------------------------
__global__ __launch_bounds__(256)
void mhdpa_qkv_ln(const float* __restrict__ x,
                  const float* __restrict__ Wq,
                  const float* __restrict__ Wk,
                  const float* __restrict__ Wv,
                  _Float16* __restrict__ qh,
                  _Float16* __restrict__ kh,
                  _Float16* __restrict__ vt) {
  __shared__ float sW[3][D_ * I_];
  for (int t = threadIdx.x; t < D_ * I_; t += 256) {
    sW[0][t] = Wq[t];
    sW[1][t] = Wk[t];
    sW[2][t] = Wv[t];
  }
  __syncthreads();

  const int gid = blockIdx.x * 256 + threadIdx.x;
  const int b = gid >> 12;        // / F_
  const int f = gid & (F_ - 1);

  float xr[D_];
#pragma unroll
  for (int d = 0; d < D_; ++d)
    xr[d] = x[((size_t)b * D_ + d) * F_ + f];

#pragma unroll
  for (int w = 0; w < 3; ++w) {
    float acc[I_];
#pragma unroll
    for (int i = 0; i < I_; ++i) acc[i] = 0.f;
    for (int d = 0; d < D_; ++d) {
      const float xv = xr[d];
      const float* wp = &sW[w][d * I_];
#pragma unroll
      for (int i = 0; i < I_; ++i) acc[i] = fmaf(xv, wp[i], acc[i]);
    }
    float mu = 0.f;
#pragma unroll
    for (int i = 0; i < I_; ++i) mu += acc[i];
    mu *= (1.0f / I_);
    float var = 0.f;
#pragma unroll
    for (int i = 0; i < I_; ++i) { float dl = acc[i] - mu; var += dl * dl; }
    var *= (1.0f / I_);
    const float rstd = rsqrtf(var + 1e-5f);

    if (w == 0) {
      _Float16* dst = qh + ((size_t)b * F_ + f) * I_;
#pragma unroll
      for (int i = 0; i < I_; ++i) dst[i] = (_Float16)((acc[i] - mu) * rstd);
    } else if (w == 1) {
      _Float16* dst = kh + ((size_t)b * F_ + f) * I_;
#pragma unroll
      for (int i = 0; i < I_; ++i) dst[i] = (_Float16)((acc[i] - mu) * rstd);
    } else {
      // v stored transposed: vt[b][i][f] so pass-B B-operands are contiguous
#pragma unroll
      for (int i = 0; i < I_; ++i)
        vt[((size_t)b * I_ + i) * F_ + f] = (_Float16)((acc[i] - mu) * rstd);
    }
  }
}

// ---------------------------------------------------------------------------
// Kernel 2: per-key-column softmax stats. One wave per (b, 2 k-tiles).
// S tile = Q(16x64) x K^T(64x16); C layout: lane holds column k=lane&15,
// rows q in the 8 accumulator VGPRs -> online max/sum over q per lane,
// halves combined with a lane-xor-16 shuffle.  Stores m and 1/Z.
// ---------------------------------------------------------------------------
__global__ __launch_bounds__(32)
void mhdpa_colstats(const _Float16* __restrict__ qh,
                    const _Float16* __restrict__ kh,
                    float* __restrict__ mOut,
                    float* __restrict__ zOut) {
  const int lane = threadIdx.x;
  const int b = blockIdx.y;
  const int k0 = blockIdx.x * 32;

  const _Float16* kr0 = kh + ((size_t)b * F_ + k0 + (lane & 15)) * I_;
  const _Float16* kr1 = kh + ((size_t)b * F_ + k0 + 16 + (lane & 15)) * I_;
  const v16h kb00 = load_B16(kr0, lane);        // i = 0..31
  const v16h kb01 = load_B16(kr0 + 32, lane);   // i = 32..63
  const v16h kb10 = load_B16(kr1, lane);
  const v16h kb11 = load_B16(kr1 + 32, lane);

  float m0 = -1e30f, z0 = 0.f, m1 = -1e30f, z1 = 0.f;

  for (int qt = 0; qt < F_; qt += 16) {
    const _Float16* qrow = qh + ((size_t)b * F_ + qt + (lane & 15)) * I_;
    const v16h a0 = load_A16(qrow, lane);
    const v16h a1 = load_A16(qrow + 32, lane);

    v8f s0 = {};
    s0 = wmma16(a0, kb00, s0);
    s0 = wmma16(a1, kb01, s0);
    v8f s1 = {};
    s1 = wmma16(a0, kb10, s1);
    s1 = wmma16(a1, kb11, s1);

    // online softmax update (per lane: one k column, 8 q rows)
    {
      float sv[8];
#pragma unroll
      for (int r = 0; r < 8; ++r) sv[r] = s0[r] * 0.125f;
      float tm = sv[0];
#pragma unroll
      for (int r = 1; r < 8; ++r) tm = fmaxf(tm, sv[r]);
      const float nm = fmaxf(m0, tm);
      float acc = z0 * __expf(m0 - nm);
#pragma unroll
      for (int r = 0; r < 8; ++r) acc += __expf(sv[r] - nm);
      z0 = acc; m0 = nm;
    }
    {
      float sv[8];
#pragma unroll
      for (int r = 0; r < 8; ++r) sv[r] = s1[r] * 0.125f;
      float tm = sv[0];
#pragma unroll
      for (int r = 1; r < 8; ++r) tm = fmaxf(tm, sv[r]);
      const float nm = fmaxf(m1, tm);
      float acc = z1 * __expf(m1 - nm);
#pragma unroll
      for (int r = 0; r < 8; ++r) acc += __expf(sv[r] - nm);
      z1 = acc; m1 = nm;
    }
  }

  // combine the two half-wave partials (rows 0-7 vs 8-15 of each tile)
  {
    const float mo = __shfl_xor(m0, 16, 32);
    const float zo = __shfl_xor(z0, 16, 32);
    const float nm = fmaxf(m0, mo);
    const float zz = z0 * __expf(m0 - nm) + zo * __expf(mo - nm);
    if (lane < 16) {
      mOut[(size_t)b * F_ + k0 + lane] = nm;
      zOut[(size_t)b * F_ + k0 + lane] = 1.0f / zz;
    }
  }
  {
    const float mo = __shfl_xor(m1, 16, 32);
    const float zo = __shfl_xor(z1, 16, 32);
    const float nm = fmaxf(m1, mo);
    const float zz = z1 * __expf(m1 - nm) + zo * __expf(mo - nm);
    if (lane < 16) {
      mOut[(size_t)b * F_ + k0 + 16 + lane] = nm;
      zOut[(size_t)b * F_ + k0 + 16 + lane] = 1.0f / zz;
    }
  }
}

// ---------------------------------------------------------------------------
// Kernel 3: out = P V. One wave per (b, 2 q-tiles); stream k in chunks of 32.
// Computes S^T = K_tile x Q^T so that the C layout of S^T (lane = fixed q,
// k across VGPRs) IS the A layout for the P@V WMMA -> no lane shuffles,
// only f32->f16 packing after exp(s-m[k])*invZ[k].
// ---------------------------------------------------------------------------
__global__ __launch_bounds__(32)
void mhdpa_attn_out(const _Float16* __restrict__ qh,
                    const _Float16* __restrict__ kh,
                    const _Float16* __restrict__ vt,
                    const float* __restrict__ mIn,
                    const float* __restrict__ zIn,
                    float* __restrict__ out) {
  const int lane = threadIdx.x;
  const int b = blockIdx.y;
  const int q0 = blockIdx.x * 32;

  // Q columns as B-operands for the S^T WMMAs (2 q-tiles x {i0..31,i32..63})
  v16h qb[2][2];
#pragma unroll
  for (int w = 0; w < 2; ++w) {
    const _Float16* qrow = qh + ((size_t)b * F_ + q0 + w * 16 + (lane & 15)) * I_;
    qb[w][0] = load_B16(qrow, lane);
    qb[w][1] = load_B16(qrow + 32, lane);
  }

  v8f co[2][4];
#pragma unroll
  for (int w = 0; w < 2; ++w)
#pragma unroll
    for (int n = 0; n < 4; ++n) co[w][n] = (v8f){};

  const float* mB = mIn + (size_t)b * F_;
  const float* zB = zIn + (size_t)b * F_;

  for (int kb = 0; kb < F_; kb += 32) {
    v16h pa[2];  // P tiles (16q x 32k) in A layout, one per q-tile
#pragma unroll
    for (int t = 0; t < 2; ++t) {
      const _Float16* krow =
          kh + ((size_t)b * F_ + kb + t * 16 + (lane & 15)) * I_;
      const v16h a0 = load_A16(krow, lane);        // K rows, i = 0..31
      const v16h a1 = load_A16(krow + 32, lane);   // K rows, i = 32..63
      const int kofs = kb + t * 16 + ((lane & 16) ? 8 : 0);
#pragma unroll
      for (int w = 0; w < 2; ++w) {
        v8f s = {};
        s = wmma16(a0, qb[w][0], s);   // S^T[k, q]
        s = wmma16(a1, qb[w][1], s);
#pragma unroll
        for (int r = 0; r < 8; ++r) {
          const float mm = mB[kofs + r];
          const float rz = zB[kofs + r];
          const float pv = __expf(s[r] * 0.125f - mm) * rz;
          pa[w][t * 8 + r] = (_Float16)pv;  // lands exactly in A layout
        }
      }
    }
    // P (16x32) x V (32x16) for each of the 4 i-subtiles
#pragma unroll
    for (int n = 0; n < 4; ++n) {
      const _Float16* vrow =
          vt + ((size_t)b * I_ + n * 16 + (lane & 15)) * F_ + kb;
      const v16h bv = load_B16(vrow, lane);
      co[0][n] = wmma16(pa[0], bv, co[0][n]);
      co[1][n] = wmma16(pa[1], bv, co[1][n]);
    }
  }

  // C layout store: lane -> i = n*16 + (lane&15); VGPR r -> q row
#pragma unroll
  for (int w = 0; w < 2; ++w)
#pragma unroll
    for (int n = 0; n < 4; ++n)
#pragma unroll
      for (int r = 0; r < 8; ++r) {
        const int q = q0 + w * 16 + r + ((lane & 16) ? 8 : 0);
        const int i = n * 16 + (lane & 15);
        out[((size_t)b * F_ + q) * I_ + i] = co[w][n][r];
      }
}

// ---------------------------------------------------------------------------
extern "C" void kernel_launch(void* const* d_in, const int* in_sizes, int n_in,
                              void* d_out, int out_size, void* d_ws,
                              size_t ws_size, hipStream_t stream) {
  (void)in_sizes; (void)n_in; (void)out_size; (void)ws_size;
  const float* x  = (const float*)d_in[0];
  const float* Wq = (const float*)d_in[1];
  const float* Wk = (const float*)d_in[2];
  const float* Wv = (const float*)d_in[3];
  float* out = (float*)d_out;

  char* ws = (char*)d_ws;
  const size_t QKV = (size_t)B_ * F_ * I_ * sizeof(_Float16);  // 8 MB each
  _Float16* qh = (_Float16*)(ws);
  _Float16* kh = (_Float16*)(ws + QKV);
  _Float16* vt = (_Float16*)(ws + 2 * QKV);
  float* mArr = (float*)(ws + 3 * QKV);
  float* zArr = (float*)(ws + 3 * QKV + (size_t)B_ * F_ * sizeof(float));

  mhdpa_qkv_ln<<<dim3((B_ * F_) / 256), dim3(256), 0, stream>>>(
      x, Wq, Wk, Wv, qh, kh, vt);
  mhdpa_colstats<<<dim3(F_ / 32, B_), dim3(32), 0, stream>>>(
      qh, kh, mArr, zArr);
  mhdpa_attn_out<<<dim3(F_ / 32, B_), dim3(32), 0, stream>>>(
      qh, kh, vt, mArr, zArr, out);
}
